// Mamba_67723044323636
// MI455X (gfx1250) — matrix operand order repaired and checked by reading, not compiled, not run on hardware
//
#include <hip/hip_runtime.h>
#include <hip/hip_bf16.h>

#define B_      2
#define L_      2048
#define D_MODEL 1024
#define D_INNER 2048
#define D_STATE 16
#define D_CONV  4
#define DT_RANK 64

typedef __attribute__((ext_vector_type(16))) __bf16 v16bf;
typedef __attribute__((ext_vector_type(8)))  __bf16 v8bf;
typedef __attribute__((ext_vector_type(8)))  float  v8f;
typedef __attribute__((ext_vector_type(4)))  unsigned int u32x4;
typedef __attribute__((ext_vector_type(8)))  int i32x8;
typedef __attribute__((ext_vector_type(4)))  int i32x4;

#if defined(__HIP_DEVICE_COMPILE__) && __has_builtin(__builtin_amdgcn_tensor_load_to_lds)
#define HAVE_TDM 1
#else
#define HAVE_TDM 0
#endif

// fp32 -> bf16 round-to-nearest-even
static __device__ __forceinline__ __bf16 f2bf(float f) {
  unsigned u = __builtin_bit_cast(unsigned, f);
  unsigned r = u + 0x7FFFu + ((u >> 16) & 1u);
  unsigned short h = (unsigned short)(r >> 16);
  return __builtin_bit_cast(__bf16, h);
}

__global__ void cvt_f32_to_bf16(const float* __restrict__ src,
                                __bf16* __restrict__ dst, int n) {
  int i = blockIdx.x * blockDim.x + threadIdx.x;
  if (i < n) dst[i] = f2bf(src[i]);
}

// ---- WMMA fragment loaders (contiguous K order k = hi*16 + s, same for A and B) ----
// A (16x32 MxK): lane l&15 = row M; 16 contiguous K values at k0 + hi*16
static __device__ __forceinline__ v16bf frag_a(const __bf16* __restrict__ A,
                                               int K, int row, int k0, int hi) {
  const __bf16* p = A + (size_t)row * K + k0 + hi * 16;
  v8bf lo = *reinterpret_cast<const v8bf*>(p);
  v8bf hh = *reinterpret_cast<const v8bf*>(p + 8);
  return __builtin_shufflevector(lo, hh, 0,1,2,3,4,5,6,7,8,9,10,11,12,13,14,15);
}
// B (32x16 KxN): lane l&15 = column N (= one K-contiguous row of W[N,K]);
// 16 contiguous K values at k0 + hi*16.
static __device__ __forceinline__ v16bf frag_b_g(const __bf16* __restrict__ W,
                                                 int K, int col, int k0, int hi) {
  return *reinterpret_cast<const v16bf*>(W + (size_t)col * K + k0 + hi * 16);
}

// ---- TDM: stage one 64(rows of W) x 64(K) bf16 tile into LDS ----
// D# per ISA 8.3/8.4: 2D tensor, data_size=2B, tile_dim0=64 (K, contiguous),
// tile_dim1=64 (rows), tensor_dim0_stride=K. Groups 2/3 zero (<=2D tensor).
static __device__ __forceinline__ void stage_b_tile(const __bf16* __restrict__ Wm,
                                                    int K, int N, int n0, int kc,
                                                    __bf16* dst, int lane) {
#if HAVE_TDM
  (void)lane;
  unsigned long long ga = (unsigned long long)(uintptr_t)(Wm + (size_t)n0 * K + kc);
  unsigned lds_addr = (unsigned)(uintptr_t)dst;     // low 32 bits = LDS byte offset
  u32x4 g0;
  g0[0] = 1u;                                       // count=1, user mode
  g0[1] = lds_addr;
  g0[2] = (unsigned)(ga & 0xFFFFFFFFu);
  g0[3] = (unsigned)((ga >> 32) & 0x1FFFFFFu) | (2u << 30);   // addr[56:32], type=2
  i32x8 g1;
  g1[0] = (int)(1u << 16);                          // workgroup_mask=0, data_size=1 (2B)
  g1[1] = (int)(((unsigned)K & 0xFFFFu) << 16);     // tensor_dim0[15:0] at bit48
  g1[2] = (int)(((unsigned)K >> 16) | (((unsigned)N & 0xFFFFu) << 16));
  g1[3] = (int)(((unsigned)N >> 16) | (64u << 16)); // tile_dim0 = 64
  g1[4] = 64;                                       // tile_dim1 = 64, tile_dim2 = 0
  g1[5] = K;                                        // tensor_dim0_stride[31:0]
  g1[6] = 0;
  g1[7] = 0;
  i32x4 z4 = {0, 0, 0, 0};
#if __clang_major__ >= 23
  i32x8 z8 = {0, 0, 0, 0, 0, 0, 0, 0};
  __builtin_amdgcn_tensor_load_to_lds(g0, g1, z4, z4, z8, 0);
#else
  __builtin_amdgcn_tensor_load_to_lds(g0, g1, z4, z4, 0);
#endif
#else
  // Fallback: wave-0 synchronous global->LDS copy (64x64 bf16 = 8KB)
  for (int it = 0; it < 16; ++it) {
    int e = it * 256 + lane * 8;
    int r = e >> 6, o = e & 63;
    *reinterpret_cast<v8bf*>(dst + r * 64 + o) =
        *reinterpret_cast<const v8bf*>(Wm + (size_t)(n0 + r) * K + kc + o);
  }
#endif
}

// ---- big WMMA GEMM: C[M,N] = A[M,K] x W[N,K]^T, fp32 accumulate ----
// Block = 4 waves; block tile 256(M) x 64(N). W tile (shared by all 4 waves)
// streamed into LDS by the Tensor Data Mover, double-buffered over K chunks
// of 64. A fragments come straight from global (L2-resident).
__global__ void __launch_bounds__(128)
gemm_wmma_bf16_big(const __bf16* __restrict__ A, const __bf16* __restrict__ Wm,
                   float* __restrict__ C, int M, int N, int K,
                   const float* __restrict__ bias, int softplus_mode) {
  __shared__ __bf16 smem[2][64 * 64];
  const int lane = threadIdx.x;          // full wave -> EXEC all ones at WMMA
  const int hi = lane >> 4, lw = lane & 15;
  const int wv = threadIdx.y;            // wave id in block (0..3)
  const int m0 = blockIdx.y * 256 + wv * 64;
  const int n0 = blockIdx.x * 64;
  const int nchunks = K >> 6;

  if (wv == 0) {
    stage_b_tile(Wm, K, N, n0, 0, smem[0], lane);   // prologue: chunk 0
  }

  v8f acc[4][4] = {};

  for (int c = 0; c < nchunks; ++c) {
    if (wv == 0) {
      if (c + 1 < nchunks) {
        stage_b_tile(Wm, K, N, n0, (c + 1) << 6, smem[(c + 1) & 1], lane);
#if HAVE_TDM
        __builtin_amdgcn_s_wait_tensorcnt(1);       // chunk c complete, c+1 in flight
#endif
      } else {
#if HAVE_TDM
        __builtin_amdgcn_s_wait_tensorcnt(0);
#endif
      }
    }
    __syncthreads();
    const __bf16* bt = smem[c & 1];
    const int kc = c << 6;
#pragma unroll
    for (int ks = 0; ks < 64; ks += 32) {
      v16bf a[4], b[4];
#pragma unroll
      for (int t = 0; t < 4; ++t)
        a[t] = frag_a(A, K, m0 + t * 16 + lw, kc + ks, hi);
#pragma unroll
      for (int t = 0; t < 4; ++t)       // ds_load_b128 pairs from staged tile
        b[t] = *reinterpret_cast<const v16bf*>(bt + (t * 16 + lw) * 64 + ks + hi * 16);
#pragma unroll
      for (int tm = 0; tm < 4; ++tm)
#pragma unroll
        for (int tn = 0; tn < 4; ++tn)
          acc[tm][tn] = __builtin_amdgcn_wmma_f32_16x16x32_bf16(
              false, a[tm], false, b[tn], (short)0, acc[tm][tn], false, false);
    }
    __syncthreads();
  }

#pragma unroll
  for (int tm = 0; tm < 4; ++tm) {
#pragma unroll
    for (int tn = 0; tn < 4; ++tn) {
      int col = n0 + tn * 16 + lw;
      float bv = softplus_mode ? bias[col] : 0.f;
#pragma unroll
      for (int j = 0; j < 8; ++j) {
        int row = m0 + tm * 16 + hi * 8 + j;       // C/D: M = hi*8 + j
        float v = acc[tm][tn][j];
        if (softplus_mode) {
          v += bv;
          v = (v > 20.f) ? v : log1pf(__expf(v));  // softplus
        }
        C[(size_t)row * N + col] = v;
      }
    }
  }
}

// ---- narrow-N WMMA GEMM (N=96 path), 32x32 per wave, direct global ----
__global__ void __launch_bounds__(128)
gemm_wmma_bf16_n32(const __bf16* __restrict__ A, const __bf16* __restrict__ W,
                   float* __restrict__ C, int M, int N, int K) {
  const int lane = threadIdx.x;
  const int hi = lane >> 4, lw = lane & 15;
  const int m0 = (blockIdx.y * blockDim.y + threadIdx.y) * 32;
  const int n0 = blockIdx.x * 32;

  v8f acc00 = {}, acc01 = {}, acc10 = {}, acc11 = {};
  for (int k0 = 0; k0 < K; k0 += 32) {
    v16bf a0 = frag_a(A, K, m0 + lw,      k0, hi);
    v16bf a1 = frag_a(A, K, m0 + 16 + lw, k0, hi);
    v16bf b0 = frag_b_g(W, K, n0 + lw,      k0, hi);
    v16bf b1 = frag_b_g(W, K, n0 + 16 + lw, k0, hi);
    acc00 = __builtin_amdgcn_wmma_f32_16x16x32_bf16(false, a0, false, b0, (short)0, acc00, false, false);
    acc01 = __builtin_amdgcn_wmma_f32_16x16x32_bf16(false, a0, false, b1, (short)0, acc01, false, false);
    acc10 = __builtin_amdgcn_wmma_f32_16x16x32_bf16(false, a1, false, b0, (short)0, acc10, false, false);
    acc11 = __builtin_amdgcn_wmma_f32_16x16x32_bf16(false, a1, false, b1, (short)0, acc11, false, false);
  }
  v8f accs[2][2] = {{acc00, acc01}, {acc10, acc11}};
#pragma unroll
  for (int tm = 0; tm < 2; ++tm)
#pragma unroll
    for (int tn = 0; tn < 2; ++tn) {
      int col = n0 + tn * 16 + lw;
#pragma unroll
      for (int j = 0; j < 8; ++j) {
        int row = m0 + tm * 16 + hi * 8 + j;
        C[(size_t)row * N + col] = accs[tm][tn][j];
      }
    }
}

// ---- depthwise causal conv (width 4) + SiLU on x-half of xz[B*L,4096] ----
__global__ void conv_silu_kernel(const float* __restrict__ xz,
                                 const float* __restrict__ conv_w,
                                 const float* __restrict__ conv_b,
                                 float* __restrict__ x_f32,
                                 __bf16* __restrict__ x_bf) {
  int idx = blockIdx.x * blockDim.x + threadIdx.x;   // over B*L*D_INNER
  if (idx >= B_ * L_ * D_INNER) return;
  int d  = idx & (D_INNER - 1);
  int bl = idx >> 11;
  int l  = bl & (L_ - 1);
  const int S = 2 * D_INNER;
  const float* base = xz + (size_t)bl * S + d;
  float acc = conv_b[d];
  float w0 = conv_w[d * 4 + 0], w1 = conv_w[d * 4 + 1];
  float w2 = conv_w[d * 4 + 2], w3 = conv_w[d * 4 + 3];
  if (l >= 3) acc += w0 * base[-3 * S];
  if (l >= 2) acc += w1 * base[-2 * S];
  if (l >= 1) acc += w2 * base[-1 * S];
  acc += w3 * base[0];
  float s = acc / (1.f + __expf(-acc));              // SiLU
  x_f32[idx] = s;
  x_bf[idx]  = f2bf(s);
}

// ---- extract dt columns (0..63) of x_dbl[B*L,96] as bf16 [B*L,64] ----
__global__ void extract_dtin(const float* __restrict__ x_dbl,
                             __bf16* __restrict__ dtin) {
  int i = blockIdx.x * blockDim.x + threadIdx.x;
  if (i >= B_ * L_ * DT_RANK) return;
  int row = i >> 6, r = i & 63;
  dtin[i] = f2bf(x_dbl[(size_t)row * 96 + r]);
}

// ---- selective scan: one thread per (b,d), 16 fp32 states in registers ----
__global__ void scan_kernel(const float* __restrict__ xz,
                            const float* __restrict__ x_f32,
                            const float* __restrict__ x_dbl,
                            const float* __restrict__ delta,
                            const float* __restrict__ A_log,
                            const float* __restrict__ Dp,
                            __bf16* __restrict__ y_bf) {
  int idx = blockIdx.x * blockDim.x + threadIdx.x;
  if (idx >= B_ * D_INNER) return;
  int b = idx >> 11;
  int d = idx & (D_INNER - 1);

  float Ad[D_STATE], h[D_STATE];
#pragma unroll
  for (int n = 0; n < D_STATE; ++n) {
    Ad[n] = -__expf(A_log[d * D_STATE + n]);
    h[n]  = 0.f;
  }
  float Dd = Dp[d];

  const size_t row0 = (size_t)b * L_;
  const float* dptr  = delta + row0 * D_INNER + d;
  const float* xptr  = x_f32 + row0 * D_INNER + d;
  const float* zptr  = xz + row0 * (2 * D_INNER) + D_INNER + d;
  const float* xdbl0 = x_dbl + row0 * 96;
  __bf16* yptr = y_bf + row0 * D_INNER + d;

  for (int l = 0; l < L_; ++l) {
    float dlt = dptr[(size_t)l * D_INNER];
    float xv  = xptr[(size_t)l * D_INNER];
    float dx  = dlt * xv;
    const float4* Bm = reinterpret_cast<const float4*>(xdbl0 + (size_t)l * 96 + DT_RANK);
    const float4* Cm = reinterpret_cast<const float4*>(xdbl0 + (size_t)l * 96 + DT_RANK + D_STATE);
    float y = 0.f;
#pragma unroll
    for (int q = 0; q < 4; ++q) {
      float4 bq = Bm[q], cq = Cm[q];
      int n = q * 4;
      h[n+0] = __expf(dlt * Ad[n+0]) * h[n+0] + dx * bq.x;  y += h[n+0] * cq.x;
      h[n+1] = __expf(dlt * Ad[n+1]) * h[n+1] + dx * bq.y;  y += h[n+1] * cq.y;
      h[n+2] = __expf(dlt * Ad[n+2]) * h[n+2] + dx * bq.z;  y += h[n+2] * cq.z;
      h[n+3] = __expf(dlt * Ad[n+3]) * h[n+3] + dx * bq.w;  y += h[n+3] * cq.w;
    }
    y += xv * Dd;
    float zv = zptr[(size_t)l * (2 * D_INNER)];
    y *= zv / (1.f + __expf(-zv));                   // * silu(z)
    yptr[(size_t)l * D_INNER] = f2bf(y);
  }
}

extern "C" void kernel_launch(void* const* d_in, const int* in_sizes, int n_in,
                              void* d_out, int out_size, void* d_ws, size_t ws_size,
                              hipStream_t stream) {
  (void)in_sizes; (void)n_in; (void)out_size; (void)ws_size;
  const float* hs     = (const float*)d_in[0];  // (B,L,D_MODEL)
  const float* in_w   = (const float*)d_in[1];  // (2*D_INNER, D_MODEL)
  const float* conv_w = (const float*)d_in[2];  // (D_INNER,1,4)
  const float* conv_b = (const float*)d_in[3];  // (D_INNER)
  const float* xp_w   = (const float*)d_in[4];  // (96, D_INNER)
  const float* dt_w   = (const float*)d_in[5];  // (D_INNER, 64)
  const float* dt_b   = (const float*)d_in[6];  // (D_INNER)
  const float* A_log  = (const float*)d_in[7];  // (D_INNER,16)
  const float* Dp     = (const float*)d_in[8];  // (D_INNER)
  const float* out_w  = (const float*)d_in[9];  // (D_MODEL, D_INNER)

  const int M = B_ * L_;                        // 4096 rows
  char* p = (char*)d_ws;
  auto take = [&](size_t bytes) { char* r = p; p += (bytes + 255) & ~(size_t)255; return r; };

  __bf16* hs_bf   = (__bf16*)take((size_t)M * D_MODEL * 2);
  __bf16* inw_bf  = (__bf16*)take((size_t)2 * D_INNER * D_MODEL * 2);
  __bf16* xpw_bf  = (__bf16*)take((size_t)96 * D_INNER * 2);
  __bf16* dtw_bf  = (__bf16*)take((size_t)D_INNER * DT_RANK * 2);
  __bf16* outw_bf = (__bf16*)take((size_t)D_MODEL * D_INNER * 2);
  float*  xz      = (float*) take((size_t)M * 2 * D_INNER * 4);
  float*  x_f32   = (float*) take((size_t)M * D_INNER * 4);
  __bf16* x_bf    = (__bf16*)take((size_t)M * D_INNER * 2);
  float*  x_dbl   = (float*) take((size_t)M * 96 * 4);
  __bf16* dtin_bf = (__bf16*)take((size_t)M * DT_RANK * 2);
  float*  delta   = (float*) take((size_t)M * D_INNER * 4);
  __bf16* y_bf    = (__bf16*)take((size_t)M * D_INNER * 2);

  auto cvt = [&](const float* s, __bf16* d, int n) {
    cvt_f32_to_bf16<<<(n + 255) / 256, 256, 0, stream>>>(s, d, n);
  };
  cvt(hs,    hs_bf,   M * D_MODEL);
  cvt(in_w,  inw_bf,  2 * D_INNER * D_MODEL);
  cvt(xp_w,  xpw_bf,  96 * D_INNER);
  cvt(dt_w,  dtw_bf,  D_INNER * DT_RANK);
  cvt(out_w, outw_bf, D_MODEL * D_INNER);

  dim3 blk(32, 4);  // 4 wave32s per block

  // GEMM1: xz[M,4096] = hs[M,1024] x in_proj_w[4096,1024]^T  (TDM-fed)
  gemm_wmma_bf16_big<<<dim3((2 * D_INNER) / 64, M / 256), blk, 0, stream>>>(
      hs_bf, inw_bf, xz, M, 2 * D_INNER, D_MODEL, nullptr, 0);

  // depthwise conv + SiLU
  {
    int n = M * D_INNER;
    conv_silu_kernel<<<(n + 255) / 256, 256, 0, stream>>>(xz, conv_w, conv_b, x_f32, x_bf);
  }

  // GEMM2: x_dbl[M,96] = x[M,2048] x x_proj_w[96,2048]^T (narrow N)
  gemm_wmma_bf16_n32<<<dim3(96 / 32, M / 128), blk, 0, stream>>>(
      x_bf, xpw_bf, x_dbl, M, 96, D_INNER);

  // dt_in -> bf16
  {
    int n = M * DT_RANK;
    extract_dtin<<<(n + 255) / 256, 256, 0, stream>>>(x_dbl, dtin_bf);
  }

  // GEMM3: delta[M,2048] = softplus(dt_in[M,64] x dt_proj_w[2048,64]^T + dt_b)
  gemm_wmma_bf16_big<<<dim3(D_INNER / 64, M / 256), blk, 0, stream>>>(
      dtin_bf, dtw_bf, delta, M, D_INNER, DT_RANK, dt_b, 1);

  // selective scan -> y_bf
  {
    int n = B_ * D_INNER;
    scan_kernel<<<(n + 255) / 256, 256, 0, stream>>>(xz, x_f32, x_dbl, delta, A_log, Dp, y_bf);
  }

  // GEMM4: out[M,1024] = y[M,2048] x out_proj_w[1024,2048]^T  (fp32 out)
  gemm_wmma_bf16_big<<<dim3(D_MODEL / 64, M / 256), blk, 0, stream>>>(
      y_bf, outw_bf, (float*)d_out, M, D_MODEL, D_INNER, nullptr, 0);
}
